// BernNet_15530601743027
// MI455X (gfx1250) — matrix operand order repaired and checked by reading, not compiled, not run on hardware
//
#include <hip/hip_runtime.h>

typedef __attribute__((ext_vector_type(16))) _Float16 v16h;
typedef __attribute__((ext_vector_type(8)))  float    v8f;

// ---------------------------------------------------------------- utilities
__global__ void zero_u32_kernel(unsigned* __restrict__ p, long long n) {
  long long i = (long long)blockIdx.x * blockDim.x + threadIdx.x;
  if (i < n) p[i] = 0u;
}

// Split W (Kdim x Ndim, row-major f32) into hi/lo f16 halves, swizzled into the
// B-fragment order used here (contiguous k, same as A): element e of lane L for (kstep, ntile) holds
// B[k = kstep*32 + (L/16)*16 + e][n = ntile*16 + (L%16)].
__global__ void swizzleB_kernel(const float* __restrict__ W,
                                _Float16* __restrict__ Bh,
                                _Float16* __restrict__ Bl,
                                int Kdim, int Ndim) {
  int idx = blockIdx.x * blockDim.x + threadIdx.x;
  int NT = Ndim >> 4;
  int total = (Kdim >> 5) * NT * 32;
  if (idx >= total) return;
  int lane  = idx & 31;
  int t2    = idx >> 5;
  int ntile = t2 % NT;
  int kstep = t2 / NT;
  int n     = ntile * 16 + (lane & 15);
  int kbase = kstep * 32 + (lane >> 4) * 16;
#pragma unroll
  for (int e = 0; e < 16; ++e) {
    float f = W[(size_t)(kbase + e) * Ndim + n];
    _Float16 h = (_Float16)f;
    Bh[(size_t)idx * 16 + e] = h;
    Bl[(size_t)idx * 16 + e] = (_Float16)(f - (float)h);
  }
}

// ------------------------------------------------------------------ GEMM
// C[Nrows,Ndim] = act(A[Nrows,Kdim] * W + bias), W pre-split/swizzled.
// block = 4 waves (128 threads); each block owns 16 rows; wave w owns N-tiles
// [w*TPW, w*TPW+TPW). fp32 accuracy via 2-term f16 split (3 WMMAs / tile / k-step).
template <int TPW>
__global__ void gemm_wmma_kernel(const float* __restrict__ A,
                                 const _Float16* __restrict__ Bh,
                                 const _Float16* __restrict__ Bl,
                                 const float* __restrict__ bias,
                                 float* __restrict__ C,
                                 int Nrows, int Kdim, int Ndim, int do_relu) {
  const int lane = threadIdx.x & 31;
  const int wave = threadIdx.x >> 5;
  const int half = lane >> 4;   // 0 or 1
  const int l16  = lane & 15;
  const int rowbase = blockIdx.x * 16;
  const int NT = Ndim >> 4;

  int arow_idx = rowbase + l16;           // A fragment: M = lane%16
  if (arow_idx >= Nrows) arow_idx = Nrows - 1;
  const float* arow = A + (size_t)arow_idx * Kdim;

  v8f zero;
#pragma unroll
  for (int e = 0; e < 8; ++e) zero[e] = 0.f;
  v8f acc[TPW];
#pragma unroll
  for (int t = 0; t < TPW; ++t) acc[t] = zero;

  for (int k0 = 0; k0 < Kdim; k0 += 32) {
    // A fragment (contiguous k order, matching B): K=k0+half*16+e
    float4 f0 = *(const float4*)(arow + k0 + half * 16);
    float4 f1 = *(const float4*)(arow + k0 + half * 16 + 4);
    float4 g0 = *(const float4*)(arow + k0 + half * 16 + 8);
    float4 g1 = *(const float4*)(arow + k0 + half * 16 + 12);
    float af[16] = {f0.x, f0.y, f0.z, f0.w, f1.x, f1.y, f1.z, f1.w,
                    g0.x, g0.y, g0.z, g0.w, g1.x, g1.y, g1.z, g1.w};
    v16h ah, al;
#pragma unroll
    for (int e = 0; e < 16; ++e) {
      _Float16 h = (_Float16)af[e];
      ah[e] = h;
      al[e] = (_Float16)(af[e] - (float)h);
    }
    int kstep = k0 >> 5;
#pragma unroll
    for (int t = 0; t < TPW; ++t) {
      int ntile = wave * TPW + t;
      size_t boff = ((size_t)(kstep * NT + ntile) * 32 + lane) * 16;
      v16h bh = *(const v16h*)(Bh + boff);
      v16h bl = *(const v16h*)(Bl + boff);
      acc[t] = __builtin_amdgcn_wmma_f32_16x16x32_f16(false, ah, false, bh,
                                                      (short)0, acc[t], false, false);
      acc[t] = __builtin_amdgcn_wmma_f32_16x16x32_f16(false, ah, false, bl,
                                                      (short)0, acc[t], false, false);
      acc[t] = __builtin_amdgcn_wmma_f32_16x16x32_f16(false, al, false, bh,
                                                      (short)0, acc[t], false, false);
    }
  }

#pragma unroll
  for (int t = 0; t < TPW; ++t) {
    int n = (wave * TPW + t) * 16 + l16;  // C/D: N = lane%16
    float b = bias[n];
#pragma unroll
    for (int r = 0; r < 8; ++r) {
      int row = rowbase + r + half * 8;   // C/D: M = r + 8*(lane/16)
      if (row < Nrows) {
        float v = acc[t][r] + b;
        if (do_relu) v = fmaxf(v, 0.f);
        C[(size_t)row * Ndim + n] = v;
      }
    }
  }
}

// ------------------------------------------------------------ graph prep
__global__ void count_kernel(const int* __restrict__ idx, unsigned* __restrict__ cnt,
                             long long E) {
  long long e = (long long)blockIdx.x * blockDim.x + threadIdx.x;
  if (e < E) atomicAdd(&cnt[idx[e]], 1u);
}

__global__ void dis_kernel(const unsigned* __restrict__ deg, float* __restrict__ dis, int n) {
  int i = blockIdx.x * blockDim.x + threadIdx.x;
  if (i < n) dis[i] = deg[i] ? rsqrtf((float)deg[i]) : 0.f;
}

// Single-block exclusive scan over per-dst counts -> row_ptr[0..n]
__global__ void scan_kernel(const unsigned* __restrict__ cnt, unsigned* __restrict__ row_ptr,
                            int n) {
  __shared__ unsigned buf[1024];
  __shared__ unsigned carry;
  int tid = threadIdx.x;
  if (tid == 0) carry = 0u;
  __syncthreads();
  for (int base = 0; base < n; base += 1024) {
    unsigned v = (base + tid < n) ? cnt[base + tid] : 0u;
    buf[tid] = v;
    __syncthreads();
    for (int off = 1; off < 1024; off <<= 1) {
      unsigned t = (tid >= off) ? buf[tid - off] : 0u;
      __syncthreads();
      buf[tid] += t;
      __syncthreads();
    }
    unsigned incl = buf[tid];
    unsigned c = carry;
    if (base + tid < n) row_ptr[base + tid] = c + incl - v;
    __syncthreads();
    if (tid == 1023) carry = c + buf[1023];
    __syncthreads();
  }
  if (tid == 0) row_ptr[n] = carry;
}

__global__ void csr_fill_kernel(const int* __restrict__ src, const int* __restrict__ dst,
                                const unsigned* __restrict__ row_ptr,
                                unsigned* __restrict__ cursor,
                                const float* __restrict__ dis,
                                int* __restrict__ col, float* __restrict__ wgt,
                                long long E) {
  long long e = (long long)blockIdx.x * blockDim.x + threadIdx.x;
  if (e >= E) return;
  int d = dst[e];
  int s = src[e];
  unsigned pos = row_ptr[d] + atomicAdd(&cursor[d], 1u);
  col[pos] = s;
  wgt[pos] = dis[s] * dis[d];
}

// alpha_j such that  sum_i coef_i*relu(temp_i)*(I-A)^i (I+A)^(K-i)  ==  sum_j alpha_j A^j
__global__ void alpha_kernel(const float* __restrict__ temp, float* __restrict__ alpha, int K) {
  if (blockIdx.x != 0 || threadIdx.x != 0) return;
  float a[32];
  for (int j = 0; j <= K; ++j) a[j] = 0.f;
  double bin = 1.0, inv2K = 1.0;
  for (int q = 0; q < K; ++q) inv2K *= 0.5;
  for (int i = 0; i <= K; ++i) {
    float t = temp[i];
    if (t < 0.f) t = 0.f;
    double c = bin * inv2K * (double)t;
    double p[33];
    for (int j = 0; j <= K + 1; ++j) p[j] = 0.0;
    p[0] = 1.0;
    for (int q = 0; q < K - i; ++q)                 // * (1+x)
      for (int j = K; j >= 1; --j) p[j] += p[j - 1];
    for (int q = 0; q < i; ++q)                     // * (1-x)
      for (int j = K; j >= 1; --j) p[j] -= p[j - 1];
    for (int j = 0; j <= K; ++j) a[j] += (float)(c * p[j]);
    bin = bin * (double)(K - i) / (double)(i + 1);
  }
  for (int j = 0; j <= K; ++j) alpha[j] = a[j];
}

__global__ void scale_kernel(const float* __restrict__ h0, const float* __restrict__ alpha,
                             int kidx, float* __restrict__ r, long long n) {
  long long i = (long long)blockIdx.x * blockDim.x + threadIdx.x;
  if (i < n) r[i] = alpha[kidx] * h0[i];
}

// rout = A*rin + alpha[j]*h0 ; one wave32 per node, lanes = 2 features each (64 total)
__global__ void spmm_kernel(const unsigned* __restrict__ row_ptr,
                            const int* __restrict__ col, const float* __restrict__ wgt,
                            const float* __restrict__ rin, const float* __restrict__ h0,
                            const float* __restrict__ alpha, int jidx,
                            float* __restrict__ rout, int n_nodes) {
  int wave = (int)((blockIdx.x * (long long)blockDim.x + threadIdx.x) >> 5);
  int lane = threadIdx.x & 31;
  if (wave >= n_nodes) return;
  unsigned s = row_ptr[wave], e = row_ptr[wave + 1];
  float ax = 0.f, ay = 0.f;
  for (unsigned base = s; base < e; base += 32u) {
    unsigned idx = base + (unsigned)lane;
    int c = 0;
    float w = 0.f;
    if (idx < e) { c = col[idx]; w = wgt[idx]; }
    unsigned cnt = e - base;
    if (cnt > 32u) cnt = 32u;
    for (unsigned i = 0; i < cnt; ++i) {
      int   cc = __shfl(c, (int)i, 32);
      float ww = __shfl(w, (int)i, 32);
      const float2 v = *(const float2*)(rin + (size_t)cc * 64 + lane * 2);
      ax += ww * v.x;
      ay += ww * v.y;
    }
  }
  float aj = alpha[jidx];
  size_t o = (size_t)wave * 64 + lane * 2;
  rout[o]     = ax + aj * h0[o];
  rout[o + 1] = ay + aj * h0[o + 1];
}

__global__ void logsoftmax_kernel(const float* __restrict__ in, float* __restrict__ out,
                                  int n_nodes) {
  int wave = (int)((blockIdx.x * (long long)blockDim.x + threadIdx.x) >> 5);
  int lane = threadIdx.x & 31;
  if (wave >= n_nodes) return;
  size_t o = (size_t)wave * 64 + lane * 2;
  float x0 = in[o], x1 = in[o + 1];
  float m = fmaxf(x0, x1);
  for (int off = 16; off; off >>= 1) m = fmaxf(m, __shfl_xor(m, off, 32));
  float s = expf(x0 - m) + expf(x1 - m);
  for (int off = 16; off; off >>= 1) s += __shfl_xor(s, off, 32);
  float l = m + logf(s);
  out[o] = x0 - l;
  out[o + 1] = x1 - l;
}

// ---------------------------------------------------------------- launcher
extern "C" void kernel_launch(void* const* d_in, const int* in_sizes, int n_in,
                              void* d_out, int out_size, void* d_ws, size_t ws_size,
                              hipStream_t stream) {
  const float* x    = (const float*)d_in[0];
  const int*   ei   = (const int*)d_in[1];
  const float* W1   = (const float*)d_in[2];
  const float* b1   = (const float*)d_in[3];
  const float* W2   = (const float*)d_in[4];
  const float* b2   = (const float*)d_in[5];
  const float* temp = (const float*)d_in[6];

  const int H  = in_sizes[3];                 // 256
  const int Cc = in_sizes[5];                 // 64
  const int F  = in_sizes[2] / H;             // 512
  const int N  = in_sizes[0] / F;             // 100000
  const long long E2 = (long long)in_sizes[1] / 2;  // 6.4M directed edges
  const int Kdeg = in_sizes[6] - 1;           // 10

  const int* srcv = ei;
  const int* dstv = ei + E2;

  char* base = (char*)d_ws;
  size_t off = 0;
  auto alloc = [&](size_t bytes) -> char* {
    char* p = base + off;
    off = (off + bytes + 255) & ~(size_t)255;
    return p;
  };
  float*    h1   = (float*)alloc((size_t)N * H * 4);
  float*    h0   = (float*)alloc((size_t)N * Cc * 4);
  float*    rA   = (float*)alloc((size_t)N * Cc * 4);
  float*    rB   = (float*)alloc((size_t)N * Cc * 4);
  _Float16* W1h  = (_Float16*)alloc((size_t)F * H * 2);
  _Float16* W1l  = (_Float16*)alloc((size_t)F * H * 2);
  _Float16* W2h  = (_Float16*)alloc((size_t)H * Cc * 2);
  _Float16* W2l  = (_Float16*)alloc((size_t)H * Cc * 2);
  unsigned* deg  = (unsigned*)alloc((size_t)N * 4);       // by src (for dis)
  unsigned* cntd = (unsigned*)alloc((size_t)N * 4);       // by dst (for CSR)
  float*    dis  = (float*)alloc((size_t)N * 4);
  unsigned* rowp = (unsigned*)alloc((size_t)(N + 1) * 4);
  unsigned* cur  = (unsigned*)alloc((size_t)N * 4);
  int*      col  = (int*)alloc((size_t)E2 * 4);
  float*    wgt  = (float*)alloc((size_t)E2 * 4);
  float*    alph = (float*)alloc(64 * 4);
  (void)ws_size; (void)n_in; (void)out_size;

  // weight split + fragment swizzle
  {
    int tot1 = (F >> 5) * (H >> 4) * 32;
    swizzleB_kernel<<<(tot1 + 127) / 128, 128, 0, stream>>>(W1, W1h, W1l, F, H);
    int tot2 = (H >> 5) * (Cc >> 4) * 32;
    swizzleB_kernel<<<(tot2 + 127) / 128, 128, 0, stream>>>(W2, W2h, W2l, H, Cc);
  }
  zero_u32_kernel<<<(N + 255) / 256, 256, 0, stream>>>(deg, N);
  zero_u32_kernel<<<(N + 255) / 256, 256, 0, stream>>>(cntd, N);
  zero_u32_kernel<<<(N + 255) / 256, 256, 0, stream>>>(cur, N);

  // dense MLP: h1 = relu(x@W1+b1) ; h0 = h1@W2+b2   (WMMA, fp32-accurate split)
  int mblocks = (N + 15) / 16;
  gemm_wmma_kernel<4><<<mblocks, 128, 0, stream>>>(x, W1h, W1l, b1, h1, N, F, H, 1);
  gemm_wmma_kernel<1><<<mblocks, 128, 0, stream>>>(h1, W2h, W2l, b2, h0, N, H, Cc, 0);

  // graph normalization + CSR (by dst)
  int eblocks = (int)((E2 + 255) / 256);
  count_kernel<<<eblocks, 256, 0, stream>>>(srcv, deg, E2);
  count_kernel<<<eblocks, 256, 0, stream>>>(dstv, cntd, E2);
  dis_kernel<<<(N + 255) / 256, 256, 0, stream>>>(deg, dis, N);
  scan_kernel<<<1, 1024, 0, stream>>>(cntd, rowp, N);
  csr_fill_kernel<<<eblocks, 256, 0, stream>>>(srcv, dstv, rowp, cur, dis, col, wgt, E2);

  // Bernstein polynomial collapsed to degree-K polynomial in A; Horner with K SpMMs
  alpha_kernel<<<1, 1, 0, stream>>>(temp, alph, Kdeg);
  long long nh = (long long)N * Cc;
  scale_kernel<<<(int)((nh + 255) / 256), 256, 0, stream>>>(h0, alph, Kdeg, rA, nh);

  int nblocks = (N + 7) / 8;  // 8 waves per 256-thread block, one wave per node
  float* cin = rA;
  float* cout = rB;
  for (int j = Kdeg - 1; j >= 0; --j) {
    spmm_kernel<<<nblocks, 256, 0, stream>>>(rowp, col, wgt, cin, h0, alph, j, cout, N);
    float* t = cin; cin = cout; cout = t;
  }

  logsoftmax_kernel<<<nblocks, 256, 0, stream>>>(cin, (float*)d_out, N);
}
